// MultiHeadedAttention_17042430231172
// MI455X (gfx1250) — compile-verified, matrix operand order checked by reading, not run on hardware
//
#include <hip/hip_runtime.h>
#include <hip/hip_bf16.h>

typedef __bf16 bf16_t;
typedef __attribute__((ext_vector_type(16))) __bf16 v16bf;
typedef __attribute__((ext_vector_type(8)))  __bf16 v8bf;
typedef __attribute__((ext_vector_type(8)))  float   v8f;
typedef __attribute__((ext_vector_type(4)))  int     v4i;

// Problem dims (fixed by the reference)
#define BATCH 64
#define UNITS 32
#define DMODEL 4096
#define CC 2        // units/16
#define DD 256      // d_model/16
#define CH 512      // CC*DD channels
#define HW 64       // 8*8 after stride-2 conv
#define DK 64       // attention head dim
#define LCOL 2048   // d_model/2
#define MROWS 2048  // BATCH*UNITS

#define WMMA_BF16(a, b, c) \
  __builtin_amdgcn_wmma_f32_16x16x32_bf16(false, (a), false, (b), (short)0, (c), false, false)

// ---- WMMA fragment loader ----------------------------------------------
// 16-bit A-matrix 16x32 layout (ISA 7.12.2): lane L holds row r0+(L&15);
// element e holds col k0 + 8*(L>=16) + 16*(e>=8) + (e&7).
// Elements 0..7 and 8..15 are each K-contiguous -> two 16-byte loads.
// Works for A ([M][K] row-major) and for B when the buffer is [N][K] row-major.
__device__ __forceinline__ v16bf load_frag_rm(const bf16_t* base, int ld, int r0, int k0) {
  int lane = threadIdx.x & 31;
  const bf16_t* p = base + (size_t)(r0 + (lane & 15)) * ld + k0 + ((lane >> 4) << 3);
  v8bf lo = *(const v8bf*)p;
  v8bf hi = *(const v8bf*)(p + 16);
  v16bf r;
#pragma unroll
  for (int i = 0; i < 8; ++i) { r[i] = lo[i]; r[i + 8] = hi[i]; }
  return r;
}

// ---- CDNA5 async global->LDS copy (ASYNCcnt path) -----------------------
// Probe-derived prototype: arg1/arg2 are generic `int4*` (global src, LDS dst).
__device__ __forceinline__ void async_copy_b128(const bf16_t* g, bf16_t* l) {
#if __has_builtin(__builtin_amdgcn_global_load_async_to_lds_b128)
  __builtin_amdgcn_global_load_async_to_lds_b128((v4i*)(void*)g, (v4i*)(void*)l, 0, 0);
#else
  *(v8bf*)l = *(const v8bf*)g;
#endif
}
__device__ __forceinline__ void async_wait0() {
#if __has_builtin(__builtin_amdgcn_s_wait_asynccnt)
  __builtin_amdgcn_s_wait_asynccnt(0);
#endif
}

// ---- 1) f32 -> bf16 convert --------------------------------------------
__global__ void k_f32_to_bf16(const float* __restrict__ src, bf16_t* __restrict__ dst, int n) {
  for (int i = blockIdx.x * blockDim.x + threadIdx.x; i < n; i += gridDim.x * blockDim.x)
    dst[i] = (bf16_t)src[i];
}

// ---- 2) patchify + depthwise conv 3x3 s2 + BN + erf-GELU ----------------
// Output transposed: yT[b][hw][ch] (bf16) so pointwise-GEMM B loads are contiguous.
__global__ void k_dwconv_bn_gelu(const float* __restrict__ x, const float* __restrict__ dw,
                                 const float* __restrict__ g, const float* __restrict__ bb,
                                 const float* __restrict__ bm, const float* __restrict__ bv,
                                 bf16_t* __restrict__ yT) {
  int i = blockIdx.x * blockDim.x + threadIdx.x;   // BATCH*CH*HW threads, exact
  int hw = i & 63;
  int ch = (i >> 6) & (CH - 1);
  int b  = i >> 15;
  int oh = hw >> 3, ow = hw & 7;
  int ci = ch >> 8, di = ch & 255;
  const float* xb = x + ((size_t)b * UNITS + ci * 16) * DMODEL + di * 16;
  const float* w  = dw + ch * 9;
  float s = 0.f;
#pragma unroll
  for (int kh = 0; kh < 3; ++kh) {
    int ih = 2 * oh - 1 + kh;
    if ((unsigned)ih < 16u) {
#pragma unroll
      for (int kw = 0; kw < 3; ++kw) {
        int iw = 2 * ow - 1 + kw;
        if ((unsigned)iw < 16u) s += w[kh * 3 + kw] * xb[ih * DMODEL + iw];
      }
    }
  }
  float yb = g[ch] * (s - bm[ch]) * rsqrtf(bv[ch] + 1e-5f) + bb[ch];
  float ge = 0.5f * yb * (1.f + erff(yb * 0.70710678118654752f));
  yT[((size_t)b * HW + hw) * CH + ch] = (bf16_t)ge;
}

// ---- 3) pointwise 1x1 conv = GEMM [512 o][512 c] x [512 c][64 hw] -------
// One wave per 16x64 strip, 4 accumulators, software-pipelined K loop.
__global__ void k_pointwise(const bf16_t* __restrict__ yT, const bf16_t* __restrict__ pw,
                            bf16_t* __restrict__ outq, bf16_t* __restrict__ vT, int want_vT) {
  int wave = (blockIdx.x * blockDim.x + threadIdx.x) >> 5;  // 2048 waves
  int lane = threadIdx.x & 31;
  int ot = wave & 31;
  int b  = wave >> 5;
  int o0 = ot * 16;
  const bf16_t* Yb = yT + (size_t)b * HW * CH;

  v8f acc[4] = {};
  v16bf a = load_frag_rm(pw, CH, o0, 0);
  v16bf bt[4];
#pragma unroll
  for (int t = 0; t < 4; ++t) bt[t] = load_frag_rm(Yb, CH, t * 16, 0);

  for (int c0 = 0; c0 < CH; c0 += 32) {
    int cn = (c0 + 32 < CH) ? c0 + 32 : 0;   // branch-free prefetch index
    v16bf an = load_frag_rm(pw, CH, o0, cn);
    v16bf btn[4];
#pragma unroll
    for (int t = 0; t < 4; ++t) btn[t] = load_frag_rm(Yb, CH, t * 16, cn);
#pragma unroll
    for (int t = 0; t < 4; ++t) acc[t] = WMMA_BF16(a, bt[t], acc[t]);
    a = an;
#pragma unroll
    for (int t = 0; t < 4; ++t) bt[t] = btn[t];
  }

  int mh = (lane >> 4) << 3;
#pragma unroll
  for (int t = 0; t < 4; ++t) {
    int n = t * 16 + (lane & 15);
#pragma unroll
    for (int vv = 0; vv < 8; ++vv) {
      int o = o0 + mh + vv;
      bf16_t val = (bf16_t)acc[t][vv];
      outq[((size_t)b * CH + o) * HW + n] = val;   // q/k/v[b][o=ci*256+di][hw]
      if (want_vT) {
        int ci = o >> 8, di = o & 255;
        vT[(((size_t)b * CC + ci) * HW + n) * DD + di] = val;  // Vt[b][ci][hw][di]
      }
    }
  }
}

// ---- 4) attention per (b, ci, 16-row tile) ------------------------------
__global__ void __launch_bounds__(256)
k_attention(const bf16_t* __restrict__ q, const bf16_t* __restrict__ k,
            const bf16_t* __restrict__ vT, const float* __restrict__ pos,
            float* __restrict__ xout) {
  __shared__ float  sc[16 * 256];     // score tile rows x all 256 cols
  __shared__ bf16_t pb[16 * 264];     // softmaxed probs, padded ld=264
  int blk = blockIdx.x;               // BATCH*CC*16
  int rt = blk & 15;
  int ci = (blk >> 4) & 1;
  int b  = blk >> 5;
  int d0 = rt * 16;
  int wid  = threadIdx.x >> 5;
  int lane = threadIdx.x & 31;
  size_t qk_base = ((size_t)b * CH + ci * DD) * DK;
  const bf16_t* Q = q + qk_base;      // [256 d][64 k]
  const bf16_t* Km = k + qk_base;     // [256 e][64 k]

  // scores = QK^T/8 + pos : each wave owns e-tiles wid and wid+8, shares Q frag
  {
    int e0a = wid * 16, e0b = (wid + 8) * 16;
    v8f acc0 = {}, acc1 = {};
#pragma unroll
    for (int k0 = 0; k0 < DK; k0 += 32) {
      v16bf a  = load_frag_rm(Q,  DK, d0, k0);
      v16bf b0 = load_frag_rm(Km, DK, e0a, k0);
      v16bf b1 = load_frag_rm(Km, DK, e0b, k0);
      acc0 = WMMA_BF16(a, b0, acc0);
      acc1 = WMMA_BF16(a, b1, acc1);
    }
    int mh = (lane >> 4) << 3;
#pragma unroll
    for (int vv = 0; vv < 8; ++vv) {
      int r = mh + vv;
      const float* pr = pos + ((size_t)ci * DD + (d0 + r)) * DD;
      sc[r * 256 + e0a + (lane & 15)] = acc0[vv] * 0.125f + pr[e0a + (lane & 15)];
      sc[r * 256 + e0b + (lane & 15)] = acc1[vv] * 0.125f + pr[e0b + (lane & 15)];
    }
  }
  __syncthreads();

  // row softmax (16 rows)
  if (threadIdx.x < 16) {
    int r = threadIdx.x;
    float mx = -3.4e38f;
    for (int e = 0; e < 256; ++e) mx = fmaxf(mx, sc[r * 256 + e]);
    float s = 0.f;
    for (int e = 0; e < 256; ++e) { float ex = __expf(sc[r * 256 + e] - mx); sc[r * 256 + e] = ex; s += ex; }
    float inv = 1.f / s;
    for (int e = 0; e < 256; ++e) pb[r * 264 + e] = (bf16_t)(sc[r * 256 + e] * inv);
  }
  __syncthreads();

  // X = P @ V : M=16, N=64 (4 waves), K=256, split-K dual accumulators
  if (wid < 4) {
    int n0 = wid * 16;
    const bf16_t* Vt = vT + ((size_t)b * CC + ci) * HW * DD;  // [hw][di]
    v8f acc0 = {}, acc1 = {};
#pragma unroll 2
    for (int k0 = 0; k0 < DD; k0 += 64) {
      v16bf a0 = load_frag_rm(pb, 264, 0, k0);
      v16bf b0 = load_frag_rm(Vt, DD, n0, k0);
      v16bf a1 = load_frag_rm(pb, 264, 0, k0 + 32);
      v16bf b1 = load_frag_rm(Vt, DD, n0, k0 + 32);
      acc0 = WMMA_BF16(a0, b0, acc0);
      acc1 = WMMA_BF16(a1, b1, acc1);
    }
    int n  = n0 + (lane & 15);
    int mh = (lane >> 4) << 3;
#pragma unroll
    for (int vv = 0; vv < 8; ++vv) {
      int r = mh + vv;
      xout[(((size_t)b * CC + ci) * DD + (d0 + r)) * DK + n] = acc0[vv] + acc1[vv];
    }
  }
}

// ---- 5) rearrange 'b c d (p1 p2) -> b (c p1) (d p2)' + conv1d(16->32) ---
__global__ void k_merge_conv1d(const float* __restrict__ x, const float* __restrict__ c1w,
                               const float* __restrict__ c1b, bf16_t* __restrict__ xm) {
  int i = blockIdx.x * blockDim.x + threadIdx.x;   // MROWS*LCOL threads, exact
  int l   = i & (LCOL - 1);
  int row = i >> 11;
  int o2 = row & 31;
  int b  = row >> 5;
  int di = l >> 3, p2 = l & 7;
  float s = c1b[o2];
#pragma unroll
  for (int ci = 0; ci < 2; ++ci)
#pragma unroll
    for (int p1 = 0; p1 < 8; ++p1)
      s += c1w[o2 * 16 + ci * 8 + p1] *
           x[(((size_t)b * CC + ci) * DD + di) * DK + p1 * 8 + p2];
  xm[(size_t)row * LCOL + l] = (bf16_t)s;
}

// ---- 6) final Linear: [2048 rows][2048 K] x [2048 K][4096 N] + bias -----
// Block = 8 waves sharing a 16-row A strip, double-buffered in LDS via async
// copies that overlap the WMMA stream. Each wave: 16x64, 4 accumulators,
// software-pipelined inner loop (A from LDS, B from L2-resident lin_w).
#define KCHUNK 512
#define NCHUNKS (LCOL / KCHUNK)
__global__ void __launch_bounds__(256)
k_final_gemm(const bf16_t* __restrict__ xm, const bf16_t* __restrict__ lw,
             const float* __restrict__ lb, float* __restrict__ out) {
  __shared__ bf16_t As[2 * 16 * KCHUNK];  // 2 x 16 KB double buffer
  int mt    = blockIdx.x >> 3;            // 128 m-tiles
  int group = blockIdx.x & 7;             // 8 n-groups of 512 cols
  int wid   = threadIdx.x >> 5;
  int lane  = threadIdx.x & 31;
  int m0    = mt * 16;
  int nbase = group * 512 + wid * 64;

  auto stage = [&](int kc, bf16_t* dst) {
#pragma unroll
    for (int j = 0; j < 4; ++j) {        // 1024 x 16B transfers over 256 threads
      int u    = j * 256 + threadIdx.x;
      int row  = u >> 6;
      int koff = (u & 63) << 3;
      async_copy_b128(xm + (size_t)(m0 + row) * LCOL + kc + koff,
                      dst + row * KCHUNK + koff);
    }
  };

  stage(0, As);
  async_wait0();
  __syncthreads();

  v8f acc[4] = {};
  for (int ic = 0; ic < NCHUNKS; ++ic) {
    int kc = ic * KCHUNK;
    const bf16_t* Ab = As + (ic & 1) * (16 * KCHUNK);
    if (ic + 1 < NCHUNKS)                 // overlap next chunk's DMA with compute
      stage(kc + KCHUNK, As + ((ic + 1) & 1) * (16 * KCHUNK));

    v16bf a = load_frag_rm(Ab, KCHUNK, 0, 0);
    v16bf bt[4];
#pragma unroll
    for (int t = 0; t < 4; ++t) bt[t] = load_frag_rm(lw, LCOL, nbase + t * 16, kc);

    for (int k0 = 0; k0 < KCHUNK; k0 += 32) {
      int kn = (k0 + 32 < KCHUNK) ? k0 + 32 : 0;   // branch-free prefetch index
      v16bf an = load_frag_rm(Ab, KCHUNK, 0, kn);
      v16bf btn[4];
#pragma unroll
      for (int t = 0; t < 4; ++t) btn[t] = load_frag_rm(lw, LCOL, nbase + t * 16, kc + kn);
#pragma unroll
      for (int t = 0; t < 4; ++t) acc[t] = WMMA_BF16(a, bt[t], acc[t]);
      a = an;
#pragma unroll
      for (int t = 0; t < 4; ++t) bt[t] = btn[t];
    }

    async_wait0();                        // next chunk's copies done
    __syncthreads();                      // all waves finished reading Ab
  }

  int mh = (lane >> 4) << 3;
#pragma unroll
  for (int t = 0; t < 4; ++t) {
    int n = nbase + t * 16 + (lane & 15);
    float bias = lb[n];
#pragma unroll
    for (int vv = 0; vv < 8; ++vv)
      out[(size_t)(m0 + mh + vv) * DMODEL + n] = acc[t][vv] + bias;
  }
}

extern "C" void kernel_launch(void* const* d_in, const int* in_sizes, int n_in,
                              void* d_out, int out_size, void* d_ws, size_t ws_size,
                              hipStream_t stream) {
  const float* q   = (const float*)d_in[0];
  const float* k   = (const float*)d_in[1];
  const float* v   = (const float*)d_in[2];
  const float* dwk = (const float*)d_in[3];
  const float* bng = (const float*)d_in[4];
  const float* bnb = (const float*)d_in[5];
  const float* bnm = (const float*)d_in[6];
  const float* bnv = (const float*)d_in[7];
  const float* pwk = (const float*)d_in[8];
  const float* pos = (const float*)d_in[9];
  const float* c1w = (const float*)d_in[10];
  const float* c1b = (const float*)d_in[11];
  const float* lw  = (const float*)d_in[12];
  const float* lb  = (const float*)d_in[13];
  float* out = (float*)d_out;

  // workspace carve-up (~56 MB)
  char* w = (char*)d_ws;
  bf16_t* pw_bf  = (bf16_t*)w;  w += (size_t)3 * 512 * 512 * sizeof(bf16_t);
  bf16_t* lw_bf  = (bf16_t*)w;  w += (size_t)4096 * 2048 * sizeof(bf16_t);
  bf16_t* yT     = (bf16_t*)w;  w += (size_t)BATCH * HW * CH * sizeof(bf16_t);
  bf16_t* qkv_bf = (bf16_t*)w;  w += (size_t)3 * BATCH * CH * HW * sizeof(bf16_t);
  bf16_t* vT     = (bf16_t*)w;  w += (size_t)BATCH * CC * HW * DD * sizeof(bf16_t);
  float*  xatt   = (float*)w;   w += (size_t)BATCH * CC * DD * DK * sizeof(float);
  bf16_t* xm     = (bf16_t*)w;  w += (size_t)MROWS * LCOL * sizeof(bf16_t);

  k_f32_to_bf16<<<3072, 256, 0, stream>>>(pwk, pw_bf, 3 * 512 * 512);
  k_f32_to_bf16<<<8192, 256, 0, stream>>>(lw, lw_bf, 4096 * 2048);

  for (int t = 0; t < 3; ++t) {
    const float* xin = (t == 0) ? q : (t == 1) ? k : v;
    k_dwconv_bn_gelu<<<8192, 256, 0, stream>>>(
        xin, dwk + (size_t)t * 512 * 9,
        bng + t * 512, bnb + t * 512, bnm + t * 512, bnv + t * 512, yT);
    k_pointwise<<<256, 256, 0, stream>>>(
        yT, pw_bf + (size_t)t * 512 * 512,
        qkv_bf + (size_t)t * BATCH * CH * HW, vT, t == 2 ? 1 : 0);
  }

  k_attention<<<2048, 256, 0, stream>>>(
      qkv_bf, qkv_bf + (size_t)BATCH * CH * HW, vT, pos, xatt);
  k_merge_conv1d<<<16384, 256, 0, stream>>>(xatt, c1w, c1b, xm);
  k_final_gemm<<<1024, 256, 0, stream>>>(xm, lw_bf, lb, out);
}
